// EuclideanEmbeddingXELoss_60559038874030
// MI455X (gfx1250) — matrix operand order repaired and checked by reading, not compiled, not run on hardware
//
#include <hip/hip_runtime.h>
#include <hip/hip_bf16.h>
#include <math.h>

typedef __attribute__((ext_vector_type(16))) __bf16 v16bf;
typedef __attribute__((ext_vector_type(8)))  __bf16 v8bf;
typedef __attribute__((ext_vector_type(8)))  float  v8f;

#define B_ROWS 4096
#define C_COLS 8192
#define D_DIM  256
#define EPS_F  1e-6f
#define DEPS2  (256.0f * (1e-6f * 1e-6f))
#define M_WG   32              // rows per workgroup (2 row-tiles of 16)

// ---------------------------------------------------------------------------
// One-shot fp32 -> bf16 hi/lo split. Each thread converts 8 consecutive
// elements (two float4 loads, two 16B bf16 stores).
// ---------------------------------------------------------------------------
__global__ void convert_kernel(const float* __restrict__ X, __bf16* __restrict__ hi,
                               __bf16* __restrict__ lo, int n_chunks) {
    int i = blockIdx.x * 256 + threadIdx.x;
    if (i >= n_chunks) return;
    size_t base = (size_t)i * 8;
    float4 f0 = ((const float4*)(X + base))[0];
    float4 f1 = ((const float4*)(X + base))[1];
    v8bf h, l;
    float xs[8] = {f0.x, f0.y, f0.z, f0.w, f1.x, f1.y, f1.z, f1.w};
    #pragma unroll
    for (int j = 0; j < 8; ++j) {
        __bf16 hh = (__bf16)xs[j];
        h[j] = hh;
        l[j] = (__bf16)(xs[j] - (float)hh);
    }
    *(v8bf*)(hi + base) = h;
    *(v8bf*)(lo + base) = l;
}

// ---------------------------------------------------------------------------
// Row-norm kernel: out[r] = sum(x^2) + sign*2*eps*sum(x), one wave per row.
// ---------------------------------------------------------------------------
__global__ void rownorm_kernel(const float* __restrict__ X, float* __restrict__ out,
                               int R, float sign) {
    int row  = blockIdx.x * 8 + (threadIdx.x >> 5);
    int lane = threadIdx.x & 31;
    if (row >= R) return;
    const float* p = X + (size_t)row * D_DIM;
    float ss = 0.f, sm = 0.f;
    #pragma unroll
    for (int d = lane; d < D_DIM; d += 32) { float x = p[d]; ss = fmaf(x, x, ss); sm += x; }
    #pragma unroll
    for (int m = 16; m; m >>= 1) { ss += __shfl_xor(ss, m, 32); sm += __shfl_xor(sm, m, 32); }
    if (lane == 0) out[row] = ss + sign * 2.0f * EPS_F * sm;
}

// ---------------------------------------------------------------------------
// Target-score kernel: ts[b] = sqrt(max(ra[b]+rc[t]+D*eps^2-2*dot(a_b,e_t),0))
// ---------------------------------------------------------------------------
__global__ void target_kernel(const float* __restrict__ A, const float* __restrict__ E,
                              const int* __restrict__ idx,
                              const float* __restrict__ ra, const float* __restrict__ rc,
                              float* __restrict__ ts) {
    int b    = blockIdx.x * 8 + (threadIdx.x >> 5);
    int lane = threadIdx.x & 31;
    if (b >= B_ROWS) return;
    int t = idx[b];
    const float* pa = A + (size_t)b * D_DIM;
    const float* pe = E + (size_t)t * D_DIM;
    float dot = 0.f;
    #pragma unroll
    for (int d = lane; d < D_DIM; d += 32) dot = fmaf(pa[d], pe[d], dot);
    #pragma unroll
    for (int m = 16; m; m >>= 1) dot += __shfl_xor(dot, m, 32);
    if (lane == 0) {
        float d2 = ra[b] + rc[t] + DEPS2 - 2.0f * dot;
        ts[b] = sqrtf(fmaxf(d2, 0.0f));
    }
}

// ---------------------------------------------------------------------------
// Main fused kernel: 32 rows per workgroup (2 row-tiles per wave), 8 waves
// sweep all 512 col-tiles. Each B fragment (hi/lo) feeds 6 WMMAs, halving
// E-side L2 traffic vs the 16-row version. Private per-lane exp-sums,
// single reduction at the end (no max-shift needed: logits <= 0).
// ---------------------------------------------------------------------------
__global__ void __launch_bounds__(256)
lse_wmma_kernel(const __bf16* __restrict__ g_ahi, const __bf16* __restrict__ g_alo,
                const __bf16* __restrict__ g_ehi, const __bf16* __restrict__ g_elo,
                const float* __restrict__ ra, const float* __restrict__ rc,
                float* __restrict__ lse) {
    __shared__ __align__(16) __bf16 a_hi[M_WG * D_DIM];
    __shared__ __align__(16) __bf16 a_lo[M_WG * D_DIM];
    __shared__ float red_s[8][M_WG];

    const int tid  = threadIdx.x;
    const int wave = tid >> 5;
    const int lane = tid & 31;
    const int row0 = blockIdx.x * M_WG;

    // Stage the 32 A rows (pre-split bf16 hi/lo) into LDS.
    {
        const v8bf* gh = (const v8bf*)(g_ahi + (size_t)row0 * D_DIM);
        const v8bf* gl = (const v8bf*)(g_alo + (size_t)row0 * D_DIM);
        v8bf* lh = (v8bf*)a_hi;
        v8bf* ll = (v8bf*)a_lo;
        #pragma unroll
        for (int i = tid; i < M_WG * D_DIM / 8; i += 256) { lh[i] = gh[i]; ll[i] = gl[i]; }
    }
    __syncthreads();

    const int h      = lane >> 4;    // accumulator half: rows r (h=0) / r+8 (h=1)
    const int n      = lane & 15;    // column-within-tile; also A row m for A frag
    const int khalf8 = h * 16;

    float rav[2][8];
    #pragma unroll
    for (int rt = 0; rt < 2; ++rt)
        #pragma unroll
        for (int r = 0; r < 8; ++r) rav[rt][r] = ra[row0 + rt * 16 + r + 8 * h];

    float st_s[2][8];
    #pragma unroll
    for (int rt = 0; rt < 2; ++rt)
        #pragma unroll
        for (int r = 0; r < 8; ++r) st_s[rt][r] = 0.0f;

    for (int ct = wave; ct < C_COLS / 16; ct += 8) {
        const int c0 = ct * 16;
        v8f acc_hh[2] = {{}, {}};
        v8f acc_hl[2] = {{}, {}};
        v8f acc_lh[2] = {{}, {}};

        #pragma unroll
        for (int k0 = 0; k0 < D_DIM; k0 += 32) {
            // B fragments (shared by both row-tiles):
            // lane: col = c0+n, elems j = e[col][k0 + 16*h + j], j = 0..15
            size_t eoff = (size_t)(c0 + n) * D_DIM + k0 + h * 16;
            v16bf bh = *(const v16bf*)(g_ehi + eoff);
            v16bf bl = *(const v16bf*)(g_elo + eoff);

            #pragma unroll
            for (int rt = 0; rt < 2; ++rt) {
                // A fragments from LDS (16-bit A 16x32 layout):
                // elems 0..15 = a[m][k0+16*h..+15] (contiguous, matches B)
                v16bf ah, al;
                const __bf16* ph = &a_hi[(rt * 16 + n) * D_DIM + k0 + khalf8];
                const __bf16* pl = &a_lo[(rt * 16 + n) * D_DIM + k0 + khalf8];
                ((v8bf*)&ah)[0] = *(const v8bf*)(ph);
                ((v8bf*)&ah)[1] = *(const v8bf*)(ph + 8);
                ((v8bf*)&al)[0] = *(const v8bf*)(pl);
                ((v8bf*)&al)[1] = *(const v8bf*)(pl + 8);

                acc_hh[rt] = __builtin_amdgcn_wmma_f32_16x16x32_bf16(
                    false, ah, false, bh, (short)0, acc_hh[rt], false, false);
                acc_hl[rt] = __builtin_amdgcn_wmma_f32_16x16x32_bf16(
                    false, ah, false, bl, (short)0, acc_hl[rt], false, false);
                acc_lh[rt] = __builtin_amdgcn_wmma_f32_16x16x32_bf16(
                    false, al, false, bh, (short)0, acc_lh[rt], false, false);
            }
        }

        // Tile epilogue: distances -> private exp-sums.
        float rc_n = rc[c0 + n];
        #pragma unroll
        for (int rt = 0; rt < 2; ++rt) {
            #pragma unroll
            for (int r = 0; r < 8; ++r) {
                float cross = acc_hh[rt][r] + acc_hl[rt][r] + acc_lh[rt][r];
                float d2 = rav[rt][r] + rc_n + DEPS2 - 2.0f * cross;
                st_s[rt][r] += __expf(-sqrtf(fmaxf(d2, 0.0f)));
            }
        }
    }

    // Reduce each slot over its 16-lane column group (once).
    #pragma unroll
    for (int rt = 0; rt < 2; ++rt) {
        #pragma unroll
        for (int r = 0; r < 8; ++r) {
            float s = st_s[rt][r];
            #pragma unroll
            for (int m = 1; m <= 8; m <<= 1) s += __shfl_xor(s, m, 32);
            st_s[rt][r] = s;
        }
    }
    if (lane == 0 || lane == 16) {
        #pragma unroll
        for (int rt = 0; rt < 2; ++rt)
            #pragma unroll
            for (int r = 0; r < 8; ++r) red_s[wave][rt * 16 + r + 8 * h] = st_s[rt][r];
    }
    __syncthreads();
    if (tid < M_WG) {
        float s = 0.0f;
        #pragma unroll
        for (int w = 0; w < 8; ++w) s += red_s[w][tid];
        lse[row0 + tid] = __logf(s);
    }
}

// ---------------------------------------------------------------------------
// Final loss: mean_b( ts[b] + lse[b] )
// ---------------------------------------------------------------------------
__global__ void loss_kernel(const float* __restrict__ ts, const float* __restrict__ lse,
                            float* __restrict__ out) {
    __shared__ float sm[256];
    float acc = 0.0f;
    for (int b = threadIdx.x; b < B_ROWS; b += 256) acc += ts[b] + lse[b];
    sm[threadIdx.x] = acc;
    __syncthreads();
    for (int s = 128; s; s >>= 1) {
        if (threadIdx.x < s) sm[threadIdx.x] += sm[threadIdx.x + s];
        __syncthreads();
    }
    if (threadIdx.x == 0) out[0] = sm[0] * (1.0f / (float)B_ROWS);
}

// ---------------------------------------------------------------------------
extern "C" void kernel_launch(void* const* d_in, const int* in_sizes, int n_in,
                              void* d_out, int out_size, void* d_ws, size_t ws_size,
                              hipStream_t stream) {
    const float* A   = (const float*)d_in[0];   // pred_embs [4096, 256]
    const int*   idx = (const int*)d_in[1];     // target_idx [4096]
    const float* E   = (const float*)d_in[2];   // all_embs [8192, 256]

    float* ws  = (float*)d_ws;
    float* ra  = ws;                            // [4096]
    float* rc  = ra + B_ROWS;                   // [8192]
    float* ts  = rc + C_COLS;                   // [4096]
    float* lse = ts + B_ROWS;                   // [4096]
    __bf16* ahi = (__bf16*)(lse + B_ROWS);      // [4096*256]
    __bf16* alo = ahi + (size_t)B_ROWS * D_DIM; // [4096*256]
    __bf16* ehi = alo + (size_t)B_ROWS * D_DIM; // [8192*256]
    __bf16* elo = ehi + (size_t)C_COLS * D_DIM; // [8192*256]  (~12.1 MB total)

    convert_kernel<<<(B_ROWS * D_DIM / 8) / 256, 256, 0, stream>>>(A, ahi, alo,
                                                                   B_ROWS * D_DIM / 8);
    convert_kernel<<<(C_COLS * D_DIM / 8) / 256, 256, 0, stream>>>(E, ehi, elo,
                                                                   C_COLS * D_DIM / 8);
    rownorm_kernel<<<B_ROWS / 8, 256, 0, stream>>>(A, ra, B_ROWS, +1.0f);
    rownorm_kernel<<<C_COLS / 8, 256, 0, stream>>>(E, rc, C_COLS, -1.0f);
    lse_wmma_kernel<<<B_ROWS / M_WG, 256, 0, stream>>>(ahi, alo, ehi, elo, ra, rc, lse);
    target_kernel<<<B_ROWS / 8, 256, 0, stream>>>(A, E, idx, ra, rc, ts);
    loss_kernel<<<1, 256, 0, stream>>>(ts, lse, (float*)d_out);
}
